// NeighbourKNN_61649960567190
// MI455X (gfx1250) — matrix operand order repaired and checked by reading, not compiled, not run on hardware
//
#include <hip/hip_runtime.h>

typedef __attribute__((ext_vector_type(16))) __bf16 v16bf;
typedef __attribute__((ext_vector_type(8)))  float  v8f;

#define NPTS   8192
#define NCH    64
#define NBATCH 4
#define KNN    16
#define COLS   16            // columns per WMMA tile step
#define WAVES_PER_BLOCK 4    // 2 row-pairs x 2 column-halves
#define ROWS_PER_PAIR   32   // two 16-row WMMA tiles per wave
#define ROWS_PER_BLOCK  64
#define THREADS         (WAVES_PER_BLOCK * 32)   // 128
#define COLHALF         (NPTS / 2)               // 4096 columns per wave

struct bfpair { __bf16 hi; __bf16 lo; };

__device__ __forceinline__ bfpair split_bf(float f) {
    bfpair r;
    __bf16 h = (__bf16)f;
    r.hi = h;
    r.lo = (__bf16)(f - (float)h);
    return r;
}

__global__ __launch_bounds__(THREADS)
void knn_topk_kernel(const float* __restrict__ x, int* __restrict__ idx_out) {
    // per-wave staging of one 32x16 key tile (padded to dodge bank conflicts)
    __shared__ float keys[WAVES_PER_BLOCK][ROWS_PER_PAIR][COLS + 1];
    // merge staging: [pair][column-half][row][k]
    __shared__ float mgd[2][2][ROWS_PER_PAIR][KNN];
    __shared__ int   mgi[2][2][ROWS_PER_PAIR][KNN];

    const int lane = threadIdx.x & 31;
    const int wave = threadIdx.x >> 5;
    const int pair = wave >> 1;     // which 32-row group
    const int half = wave & 1;      // which half of the candidate columns
    const int h    = lane >> 4;     // half-wave select
    const int m    = lane & 15;

    const int blocks_per_batch = NPTS / ROWS_PER_BLOCK;  // 128
    const int batch   = blockIdx.x / blocks_per_batch;
    const int rowbase = (blockIdx.x % blocks_per_batch) * ROWS_PER_BLOCK
                        + pair * ROWS_PER_PAIR;
    const float* xb = x + (size_t)batch * NPTS * NCH;

    // ---------- preload A fragments: 2 row-tiles x 2 K-chunks, bf16 hi/lo split
    // A 16x32 bf16 layout: lanes0-15 (M=lane): v0..v3 = K 0..7 (pairs),
    // v4..v7 = K 16..23 ; lanes16-31: K 8..15 / 24..31.
    v16bf a_hi[2][2], a_lo[2][2];
    #pragma unroll
    for (int t = 0; t < 2; ++t) {
        const int row = rowbase + t * 16 + m;
        const float2* rp = (const float2*)(xb + (size_t)row * NCH);
        #pragma unroll
        for (int c = 0; c < 2; ++c) {
            const int p0 = (c * 32 + 8 * h) >> 1;       // float2 idx, K-low group
            const int p1 = (c * 32 + 16 + 8 * h) >> 1;  // float2 idx, K-high group
            #pragma unroll
            for (int i = 0; i < 4; ++i) {
                float2 f0 = rp[p0 + i];
                float2 f1 = rp[p1 + i];
                bfpair s;
                s = split_bf(f0.x); a_hi[t][c][2*i]         = s.hi; a_lo[t][c][2*i]         = s.lo;
                s = split_bf(f0.y); a_hi[t][c][2*i + 1]     = s.hi; a_lo[t][c][2*i + 1]     = s.lo;
                s = split_bf(f1.x); a_hi[t][c][8 + 2*i]     = s.hi; a_lo[t][c][8 + 2*i]     = s.lo;
                s = split_bf(f1.y); a_hi[t][c][8 + 2*i + 1] = s.hi; a_lo[t][c][8 + 2*i + 1] = s.lo;
            }
        }
    }

    // ---------- top-16 state, sorted ascending (bd[15] = current worst)
    float bd[KNN];
    int   bi[KNN];
    #pragma unroll
    for (int j = 0; j < KNN; ++j) { bd[j] = 3.0e38f; bi[j] = 0; }

    // ---------- stream over this wave's half of the candidate columns
    const int colstart = half * COLHALF;
    const int colend   = colstart + COLHALF;
    for (int colbase = colstart; colbase < colend; colbase += COLS) {
        // B 32x16 bf16 layout: lane holds column (lane&15); lanes0-15 carry
        // K 0..7 / 16..23 (pairs per VGPR), lanes16-31 carry K 8..15 / 24..31.
        const int col = colbase + m;
        const float2* cp = (const float2*)(xb + (size_t)col * NCH);

        // prefetch next column tile (L2-resident)
        if (colbase + COLS < colend)
            __builtin_prefetch(xb + (size_t)(col + COLS) * NCH, 0, 3);

        v16bf b_hi[2], b_lo[2];
        float part = 0.0f;                     // partial ||col||^2 (this lane's K half)
        #pragma unroll
        for (int c = 0; c < 2; ++c) {
            const int pk = (c * 32 + 8 * h) >> 1;    // float2 index of K-low group start
            #pragma unroll
            for (int i = 0; i < 8; ++i) {
                float2 f = cp[pk + i + (i & 4)];
                part += f.x * f.x + f.y * f.y;
                bfpair s;
                s = split_bf(f.x); b_hi[c][2*i]     = s.hi; b_lo[c][2*i]     = s.lo;
                s = split_bf(f.y); b_hi[c][2*i + 1] = s.hi; b_lo[c][2*i + 1] = s.lo;
            }
        }
        // complete the column norm across the two half-waves (wave32 xor-shuffle)
        const float xxc = part + __shfl_xor(part, 16, 32);

        // ---------- 16x16x64 inner products via bf16-split WMMA (3 terms/chunk)
        v8f acc0 = {}, acc1 = {};
        #pragma unroll
        for (int c = 0; c < 2; ++c) {
            acc0 = __builtin_amdgcn_wmma_f32_16x16x32_bf16(false, a_hi[0][c], false, b_hi[c], (short)0, acc0, false, false);
            acc0 = __builtin_amdgcn_wmma_f32_16x16x32_bf16(false, a_hi[0][c], false, b_lo[c], (short)0, acc0, false, false);
            acc0 = __builtin_amdgcn_wmma_f32_16x16x32_bf16(false, a_lo[0][c], false, b_hi[c], (short)0, acc0, false, false);
            acc1 = __builtin_amdgcn_wmma_f32_16x16x32_bf16(false, a_hi[1][c], false, b_hi[c], (short)0, acc1, false, false);
            acc1 = __builtin_amdgcn_wmma_f32_16x16x32_bf16(false, a_hi[1][c], false, b_lo[c], (short)0, acc1, false, false);
            acc1 = __builtin_amdgcn_wmma_f32_16x16x32_bf16(false, a_lo[1][c], false, b_hi[c], (short)0, acc1, false, false);
        }

        // ---------- stage keys (||b||^2 - 2<a,b>; row norm is ordering-invariant)
        __syncthreads();   // previous iteration's reads are done
        #pragma unroll
        for (int v = 0; v < 8; ++v) {
            const int r = v + 8 * h;           // C/D layout: lanes16-31 -> M = v+8
            keys[wave][r][m]      = xxc - 2.0f * acc0[v];
            keys[wave][16 + r][m] = xxc - 2.0f * acc1[v];
        }
        __syncthreads();   // keys visible

        // ---------- per-lane top-16: lane owns local row == lane
        #pragma unroll
        for (int n = 0; n < COLS; ++n) {
            const float d = keys[wave][lane][n];
            if (d < bd[KNN - 1]) {             // rare after warmup
                const int ci = colbase + n;
                #pragma unroll
                for (int j = KNN - 1; j > 0; --j) {
                    if (bd[j] > d) {
                        const bool sh = bd[j - 1] > d;
                        bd[j] = sh ? bd[j - 1] : d;
                        bi[j] = sh ? bi[j - 1] : ci;
                    }
                }
                if (bd[0] > d) { bd[0] = d; bi[0] = ci; }
            }
        }
    }

    // ---------- merge the two column-halves' sorted lists per row
    __syncthreads();
    #pragma unroll
    for (int j = 0; j < KNN; ++j) {
        mgd[pair][half][lane][j] = bd[j];
        mgi[pair][half][lane][j] = bi[j];
    }
    __syncthreads();

    if (half == 0) {
        const float* da = mgd[pair][0][lane];
        const int*   ia = mgi[pair][0][lane];
        const float* db = mgd[pair][1][lane];
        const int*   ib = mgi[pair][1][lane];
        const size_t orow = ((size_t)batch * NPTS + rowbase + lane) * KNN;
        int pa = 0, pb = 0;
        for (int j = 0; j < KNN; ++j) {
            const float va = da[pa];
            const float vb = db[pb];
            const bool takeA = va <= vb;       // tie -> lower column index (half 0)
            idx_out[orow + j] = takeA ? ia[pa] : ib[pb];
            pa += takeA ? 1 : 0;
            pb += takeA ? 0 : 1;
        }
    }
}

extern "C" void kernel_launch(void* const* d_in, const int* in_sizes, int n_in,
                              void* d_out, int out_size, void* d_ws, size_t ws_size,
                              hipStream_t stream) {
    (void)in_sizes; (void)n_in; (void)out_size; (void)d_ws; (void)ws_size;
    const float* x = (const float*)d_in[0];

    // output 0: x passed through
    const size_t xelems = (size_t)NBATCH * NPTS * NCH;
    (void)hipMemcpyAsync(d_out, x, xelems * sizeof(float),
                         hipMemcpyDeviceToDevice, stream);

    // output 1: int32 knn indices, concatenated after x
    int* idx_out = (int*)((float*)d_out + xelems);

    dim3 grid(NBATCH * (NPTS / ROWS_PER_BLOCK));   // 512 blocks
    dim3 block(THREADS);                           // 128 threads = 4 waves
    knn_topk_kernel<<<grid, block, 0, stream>>>(x, idx_out);
}
